// TTT_5257039970500
// MI455X (gfx1250) — matrix operand order repaired and checked by reading, not compiled, not run on hardware
//
#include <hip/hip_runtime.h>
#include <hip/hip_bf16.h>
#include <cstdint>
#include <cstddef>

// ---------------- fixed problem dims (from reference) ----------------
#define BB    2
#define TT    4096
#define DD    2048
#define HH    16
#define DH    128          // head dim
#define CH    16           // TTT chunk
#define NCHK  (TT/CH)      // 256 chunks
#define MROWS (BB*TT)      // 8192
#define EPSF  1e-6f

typedef __bf16 bf16_t;
typedef __attribute__((ext_vector_type(16))) __bf16 v16bf;
typedef __attribute__((ext_vector_type(8)))  float  v8f;
typedef __attribute__((ext_vector_type(4)))  unsigned int v4u;
typedef __attribute__((ext_vector_type(8)))  int v8i;
typedef __attribute__((ext_vector_type(4)))  int v4i;

static __device__ __forceinline__ float sigm_(float x) {
    return __builtin_amdgcn_rcpf(1.f + __expf(-x));   // v_rcp_f32, not IEEE divide
}

// Async global->LDS 16B per lane (CDNA5, ASYNCcnt-tracked). INST_OFFSET is
// applied to BOTH the LDS and global addresses (ISA 15.18.3 pseudocode).
static __device__ __forceinline__ void async_b128(unsigned ldsaddr, const void* gaddr) {
    asm volatile("global_load_async_to_lds_b128 %0, %1, off"
                 :: "v"(ldsaddr), "v"(gaddr) : "memory");
}
static __device__ __forceinline__ void async_b128_off16(unsigned ldsaddr, const void* gaddr) {
    asm volatile("global_load_async_to_lds_b128 %0, %1, off offset:16"
                 :: "v"(ldsaddr), "v"(gaddr) : "memory");
}
static __device__ __forceinline__ void wait_async0() {
    asm volatile("s_wait_asynccnt 0x0" ::: "memory");
}
static __device__ __forceinline__ unsigned lds_addr(const void* p) {
    return (unsigned)(uintptr_t)p;   // LDS aperture: byte offset lives in addr[31:0]
}

// TDM: issue a 2D-tile tensor DMA  global[16 x 128 f32, row stride 2048] -> LDS
// (packed). D# bitfields per CDNA5 ISA ch.8. Call from ONE wave (EXEC ignored,
// one DMA per instruction); completion via s_wait_tensorcnt.
static __device__ __forceinline__ void tdm_load_tile_16x128_f32(
    unsigned ldsaddr, const float* gptr)
{
    unsigned long long ga = (unsigned long long)(uintptr_t)gptr;
    // group0: count=1 | lds_addr | global_addr[56:0] | type=2 (bits 127:126)
    v4u g0 = { 1u, ldsaddr, (unsigned)ga,
               (unsigned)((ga >> 32) & 0x01FFFFFFu) | 0x80000000u };
    // group1: data_size=4B (bits 17:16); tensor_dim0=2048 (bits 79:48);
    // tensor_dim1=8192 (bits 111:80); tile_dim0=128 (127:112); tile_dim1=16
    // (143:128); tensor_dim0_stride=2048 (207:160); rest 0.
    v8i g1 = { (int)0x00020000,            // data_size=2 (4 bytes)
               (int)0x08000000,            // tensor_dim0[15:0]=2048 in [31:16]
               (int)0x20000000,            // tensor_dim1[15:0]=8192 in [31:16]
               (int)0x00800000,            // tile_dim0=128 in [31:16]
               (int)16,                    // tile_dim1=16
               (int)2048,                  // tensor_dim0_stride lo
               0, 0 };
    v4i gz4 = { 0, 0, 0, 0 };              // 2D tensor: groups 2/3 unused
    v8i gz8 = { 0, 0, 0, 0, 0, 0, 0, 0 };
    __builtin_amdgcn_tensor_load_to_lds(g0, g1, gz4, gz4, gz8, 0);
}

// =====================================================================
// fp32 -> bf16 conversion (flat)
// =====================================================================
__global__ void cvt_f32_to_bf16(const float* __restrict__ in, bf16_t* __restrict__ out, int n) {
    int i = (blockIdx.x * blockDim.x + threadIdx.x) * 4;
    if (i >= n) return;
    if (i + 3 < n) {
        float4 f = *(const float4*)(in + i);
        out[i + 0] = (bf16_t)f.x;
        out[i + 1] = (bf16_t)f.y;
        out[i + 2] = (bf16_t)f.z;
        out[i + 3] = (bf16_t)f.w;
    } else {
        for (int j = i; j < n; ++j) out[j] = (bf16_t)in[j];
    }
}

// =====================================================================
// fp32 [K][N] -> bf16 transposed [N][K] (tiled via LDS)
// =====================================================================
__global__ __launch_bounds__(256) void cvt_transpose_kernel(
    const float* __restrict__ in, bf16_t* __restrict__ out, int K, int N)
{
    __shared__ float tile[32][33];
    const int n0 = blockIdx.x * 32, k0 = blockIdx.y * 32;
    const int tx = threadIdx.x, ty = threadIdx.y;   // 32 x 8
    #pragma unroll
    for (int i = 0; i < 32; i += 8)
        tile[ty + i][tx] = in[(size_t)(k0 + ty + i) * N + n0 + tx];   // tile[k][n]
    __syncthreads();
    #pragma unroll
    for (int i = 0; i < 32; i += 8)
        out[(size_t)(n0 + ty + i) * K + k0 + tx] = (bf16_t)tile[tx][ty + i];
}

// =====================================================================
// bf16 WMMA GEMM: C[M,N] (f32) = act( A[M,K] @ B[K,N] ), B given TRANSPOSED
// as BT[N][K]. Block tile 128x128, K-step 32. 8 waves: 4(M) x 2(N).
// Double-buffered LDS filled by global_load_async_to_lds_b128.
// act: 0 = none, 1 = silu
// =====================================================================
#define GTM 128
#define GTN 128
#define GTK 32
__global__ __launch_bounds__(256) void gemm_bf16_kernel(
    const bf16_t* __restrict__ A, const bf16_t* __restrict__ BT, float* __restrict__ C,
    int M, int N, int K, int act)
{
    __shared__ bf16_t As [2][GTM][GTK + 8];   // 2 x 128 x 40 bf16
    __shared__ bf16_t BsT[2][GTN][GTK + 8];   // 2 x 128 x 40 bf16 (transposed B)

    const int tid  = threadIdx.x;
    const int lane = tid & 31;
    const int wid  = tid >> 5;
    const int bm = blockIdx.y * GTM;
    const int bn = blockIdx.x * GTN;
    const int wm = (wid >> 1) * 32;       // 0,32,64,96
    const int wn = (wid & 1) * 64;        // 0,64

    const int lM    = lane & 15;
    const int khalf = (lane & 16) ? 8 : 0;
    const int nN    = lane & 15;
    const int koff  = (lane & 16) ? 16 : 0;
    const int mh    = (lane & 16) ? 8 : 0;

    v8f acc[2][4] = {};

    const int lr = tid >> 1, lc = (tid & 1) * 16;   // tile-stage coords (rows 0..127)

    auto stage_async = [&](int buf, int k0) {
        const bf16_t* srcA = A  + (size_t)(bm + lr) * K + k0 + lc;
        unsigned dstA = lds_addr(&As[buf][lr][lc]);
        async_b128(dstA, srcA);
        async_b128_off16(dstA, srcA);
        const bf16_t* srcB = BT + (size_t)(bn + lr) * K + k0 + lc;
        unsigned dstB = lds_addr(&BsT[buf][lr][lc]);
        async_b128(dstB, srcB);
        async_b128_off16(dstB, srcB);
    };

    stage_async(0, 0);
    wait_async0();
    __syncthreads();

    int buf = 0;
    for (int k0 = 0; k0 < K; k0 += GTK) {
        const int nxt = k0 + GTK;
        if (nxt < K) stage_async(buf ^ 1, nxt);   // async fill of next tile

        // ---- fragments (contiguous LDS runs -> ds_load_b128) ----
        v16bf afr[2], bfr[4];
        #pragma unroll
        for (int i = 0; i < 2; ++i) {
            const int R = wm + i * 16;
            #pragma unroll
            for (int e = 0; e < 16; ++e) {
                int kk = e + koff;   // one contiguous 16-run
                afr[i][e] = As[buf][R + lM][kk];
            }
        }
        #pragma unroll
        for (int j = 0; j < 4; ++j) {
            const int Cb = wn + j * 16;
            #pragma unroll
            for (int e = 0; e < 16; ++e)
                bfr[j][e] = BsT[buf][Cb + nN][koff + e]; // 16 contiguous bf16
        }
        #pragma unroll
        for (int i = 0; i < 2; ++i)
            #pragma unroll
            for (int j = 0; j < 4; ++j)
                acc[i][j] = __builtin_amdgcn_wmma_f32_16x16x32_bf16(
                    false, afr[i], false, bfr[j], (short)0, acc[i][j], false, false);

        if (nxt < K) wait_async0();               // my async writes landed
        __syncthreads();                          // everyone done reading buf / writing buf^1
        buf ^= 1;
    }

    // ---- epilogue ----
    #pragma unroll
    for (int i = 0; i < 2; ++i)
        #pragma unroll
        for (int j = 0; j < 4; ++j)
            #pragma unroll
            for (int r = 0; r < 8; ++r) {
                int row = bm + wm + i * 16 + r + mh;
                int col = bn + wn + j * 16 + nN;
                float val = acc[i][j][r];
                if (act == 1) val = val * sigm_(val);   // silu
                C[(size_t)row * N + col] = val;
            }
}

// =====================================================================
// per-(row,head) L2 normalize of k (in place). One wave per (row,head).
// =====================================================================
__global__ __launch_bounds__(256) void l2norm_heads_kernel(float* __restrict__ kf) {
    int gw   = blockIdx.x * 8 + (threadIdx.x >> 5);
    int lane = threadIdx.x & 31;
    int row = gw / HH, h = gw % HH;
    if (row >= MROWS) return;
    float* p = kf + (size_t)row * DD + h * DH + lane * 4;
    float4 x = *(float4*)p;
    float ss = x.x * x.x + x.y * x.y + x.z * x.z + x.w * x.w;
    #pragma unroll
    for (int m = 16; m >= 1; m >>= 1) ss += __shfl_xor(ss, m, 32);
    float r = rsqrtf(ss + EPSF);
    x.x *= r; x.y *= r; x.z *= r; x.w *= r;
    *(float4*)p = x;
}

// =====================================================================
// eta = 2*sigmoid(x @ Wbeta). One wave per row; 16 head accumulators.
// =====================================================================
__global__ __launch_bounds__(256) void beta_kernel(
    const float* __restrict__ x, const float* __restrict__ wbeta, float* __restrict__ etaf)
{
    int row  = blockIdx.x * 8 + (threadIdx.x >> 5);
    int lane = threadIdx.x & 31;
    if (row >= MROWS) return;
    float acc[HH];
    #pragma unroll
    for (int h = 0; h < HH; ++h) acc[h] = 0.f;
    for (int kk = lane; kk < DD; kk += 32) {
        float xv = x[(size_t)row * DD + kk];
        #pragma unroll
        for (int h = 0; h < HH; ++h) acc[h] += xv * wbeta[(size_t)kk * HH + h];
    }
    #pragma unroll
    for (int h = 0; h < HH; ++h) {
        float s = acc[h];
        #pragma unroll
        for (int m = 16; m >= 1; m >>= 1) s += __shfl_xor(s, m, 32);
        if (lane == 0) etaf[(size_t)row * HH + h] = 2.f * sigm_(s);
    }
}

// =====================================================================
// TTT-linear chunked dual-form scan. One block per (b,h); 8 waves.
// State S kept TRANSPOSED in LDS: St[e][d] = S[d][e]  (f32, resident).
// Chunk q/k tiles staged by the Tensor Data Mover (tensor_load_to_lds,
// TENSORcnt). All LDS fragment walks contiguous; matmuls via v_wmma.
// =====================================================================
__global__ __launch_bounds__(256) void ttt_scan_kernel(
    const float* __restrict__ qf, const float* __restrict__ kf, const float* __restrict__ vf,
    const float* __restrict__ etaf, const float* __restrict__ lnw_g, const float* __restrict__ lnb_g,
    float* __restrict__ of)
{
    __shared__ float St[DH][DH + 4];     // S transposed: St[e][d], 67.6 KB
    __shared__ float Qc[CH][DH];         // chunk q  (TDM dest, packed 16x128)
    __shared__ float Kc[CH][DH];         // chunk k  (TDM dest, packed 16x128)
    __shared__ float KH[CH][DH];         // kh then kh_hat          (8 KB)
    __shared__ float GW[CH][DH];         // gw                      (8 KB)
    __shared__ float PR[CH][DH];         // gw*kh_hat, then qh      (8 KB)
    __shared__ float VNT[DH][CH];        // v_new transposed [d][c] (8 KB)
    __shared__ float KET[DH][CH];        // (k*eta)^T        [d][c] (8 KB)
    __shared__ float Wln[DH], Bln[DH];
    __shared__ float Ei[CH];
    __shared__ float Amat[CH][CH];
    __shared__ float muS[CH], rsS[CH], m1S[CH], m2S[CH];

    const int tid  = threadIdx.x;
    const int lane = tid & 31;
    const int wid  = tid >> 5;           // 0..7
    const int cb   = wid * 16;           // wave-private S column slab
    const int b = blockIdx.x / HH;
    const int h = blockIdx.x % HH;

    const int lM    = lane & 15;
    const int khalf = (lane & 16) ? 8 : 0;
    const int nN    = lane & 15;
    const int koff  = (lane & 16) ? 16 : 0;
    const int mh    = (lane & 16) ? 8 : 0;

    for (int i = tid; i < DH; i += 256) { Wln[i] = lnw_g[h * DH + i]; Bln[i] = lnb_g[h * DH + i]; }
    for (int i = tid; i < DH * (DH + 4); i += 256) (&St[0][0])[i] = 0.f;
    __syncthreads();

    for (int nc = 0; nc < NCHK; ++nc) {
        const int rowbase = b * TT + nc * CH;

        // ---- TDM: DMA q,k chunk tiles into LDS (wave 0 issues, one DMA each) ----
        if (wid == 0) {
            tdm_load_tile_16x128_f32(lds_addr(&Qc[0][0]),
                                     qf + (size_t)rowbase * DD + h * DH);
            tdm_load_tile_16x128_f32(lds_addr(&Kc[0][0]),
                                     kf + (size_t)rowbase * DD + h * DH);
            __builtin_amdgcn_s_wait_tensorcnt(0);
        }
        if (tid < CH) Ei[tid] = etaf[(size_t)(rowbase + tid) * HH + h];
        __syncthreads();

        // ---- P1: kh slab = Kc[16x128] @ S[:, cb:cb+16]  (B from St rows) ----
        {
            v8f acc = {};
            #pragma unroll
            for (int kb = 0; kb < DH; kb += 32) {
                v16bf a, bb;
                #pragma unroll
                for (int e = 0; e < 16; ++e) {
                    int kk = kb + e + koff;
                    a[e] = (bf16_t)Kc[lM][kk];
                }
                #pragma unroll
                for (int e = 0; e < 16; ++e)
                    bb[e] = (bf16_t)St[cb + nN][kb + koff + e];   // contiguous
                acc = __builtin_amdgcn_wmma_f32_16x16x32_bf16(false, a, false, bb, (short)0, acc, false, false);
            }
            #pragma unroll
            for (int r = 0; r < 8; ++r) KH[r + mh][cb + nN] = acc[r];
        }
        __syncthreads();

        // ---- P2a: LN stats of kh rows ----
        if (tid < CH) {
            float s = 0.f, s2 = 0.f;
            for (int c = 0; c < DH; c += 4) {
                float4 xv = *(const float4*)&KH[tid][c];
                s  += xv.x + xv.y + xv.z + xv.w;
                s2 += xv.x*xv.x + xv.y*xv.y + xv.z*xv.z + xv.w*xv.w;
            }
            float mu = s / DH;
            muS[tid] = mu;
            rsS[tid] = rsqrtf(s2 / DH - mu * mu + EPSF);
        }
        __syncthreads();
        // ---- P2b: kh_hat, g, gw, gw*kh_hat; also build KET = (Kc*eta)^T ----
        {
            int idx = tid * 8;
            #pragma unroll
            for (int i = 0; i < 8; ++i) {
                int r = (idx + i) >> 7, c = (idx + i) & 127;
                float khh = (KH[r][c] - muS[r]) * rsS[r];
                KH[r][c] = khh;
                float tgt = vf[(size_t)(rowbase + r) * DD + h * DH + c] - Kc[r][c];
                float g  = Wln[c] * khh + Bln[c] - tgt;
                float gw = g * Wln[c];
                GW[r][c] = gw;
                PR[r][c] = gw * khh;
            }
            // KET[d][c] = Kc[c][d] * Ei[c]   (128x16)
            int d = tid >> 1, c0 = (tid & 1) * 8;
            #pragma unroll
            for (int i = 0; i < 8; ++i)
                KET[d][c0 + i] = Kc[c0 + i][d] * Ei[c0 + i];
        }
        __syncthreads();
        // ---- P2c: row means of gw and gw*kh_hat ----
        if (tid < CH) {
            float s1 = 0.f, s2 = 0.f;
            for (int c = 0; c < DH; c += 4) {
                float4 a4 = *(const float4*)&GW[tid][c];
                float4 b4 = *(const float4*)&PR[tid][c];
                s1 += a4.x + a4.y + a4.z + a4.w;
                s2 += b4.x + b4.y + b4.z + b4.w;
            }
            m1S[tid] = s1 / DH; m2S[tid] = s2 / DH;
        }
        __syncthreads();
        // ---- P2d: v_new -> VNT (transposed) ----
        {
            int idx = tid * 8;
            #pragma unroll
            for (int i = 0; i < 8; ++i) {
                int r = (idx + i) >> 7, c = (idx + i) & 127;
                VNT[c][r] = (GW[r][c] - m1S[r] - KH[r][c] * m2S[r]) * rsS[r];
            }
        }
        __syncthreads();

        // ---- P3: A = tril(Qc @ Kc^T) * eta[s]  (wave 0) ----
        if (wid == 0) {
            v8f acc = {};
            #pragma unroll
            for (int kb = 0; kb < DH; kb += 32) {
                v16bf a, bb;
                #pragma unroll
                for (int e = 0; e < 16; ++e) {
                    int kk = kb + e + koff;
                    a[e] = (bf16_t)Qc[lM][kk];
                }
                #pragma unroll
                for (int e = 0; e < 16; ++e)
                    bb[e] = (bf16_t)Kc[nN][kb + koff + e];   // row of Kc: contiguous
                acc = __builtin_amdgcn_wmma_f32_16x16x32_bf16(false, a, false, bb, (short)0, acc, false, false);
            }
            #pragma unroll
            for (int r = 0; r < 8; ++r) {
                int c = r + mh, s = nN;
                Amat[c][s] = (s <= c) ? acc[r] * Ei[s] : 0.f;
            }
        }
        __syncthreads();

        // ---- P4: qh slab = Qc @ S[:,slab] - Amat @ v_new[:,slab] ----
        {
            v8f acc = {};
            #pragma unroll
            for (int kb = 0; kb < DH; kb += 32) {
                v16bf a, bb;
                #pragma unroll
                for (int e = 0; e < 16; ++e) {
                    int kk = kb + e + koff;
                    a[e] = (bf16_t)Qc[lM][kk];
                }
                #pragma unroll
                for (int e = 0; e < 16; ++e)
                    bb[e] = (bf16_t)St[cb + nN][kb + koff + e];
                acc = __builtin_amdgcn_wmma_f32_16x16x32_bf16(false, a, false, bb, (short)0, acc, false, false);
            }
            v16bf aa, bb2;
            #pragma unroll
            for (int e = 0; e < 16; ++e) {
                int kk = e + koff;
                aa[e] = (kk < 16) ? (bf16_t)Amat[lM][kk] : (bf16_t)0.0f;
            }
            #pragma unroll
            for (int e = 0; e < 16; ++e)   // lanes<16: VNT row (contiguous); lanes>=16: zero pad
                bb2[e] = (koff == 0) ? (bf16_t)VNT[cb + nN][e] : (bf16_t)0.0f;
            v8f neg = {};
            neg = __builtin_amdgcn_wmma_f32_16x16x32_bf16(false, aa, false, bb2, (short)0, neg, false, false);
            #pragma unroll
            for (int r = 0; r < 8; ++r) PR[r + mh][cb + nN] = acc[r] - neg[r];
        }
        __syncthreads();

        // ---- P5a: LN stats of qh rows ----
        if (tid < CH) {
            float s = 0.f, s2 = 0.f;
            for (int c = 0; c < DH; c += 4) {
                float4 xv = *(const float4*)&PR[tid][c];
                s  += xv.x + xv.y + xv.z + xv.w;
                s2 += xv.x*xv.x + xv.y*xv.y + xv.z*xv.z + xv.w*xv.w;
            }
            float mu = s / DH;
            muS[tid] = mu;
            rsS[tid] = rsqrtf(s2 / DH - mu * mu + EPSF);
        }
        __syncthreads();
        // ---- P5b: o = q + LN(qh) -> global ----
        {
            int idx = tid * 8;
            #pragma unroll
            for (int i = 0; i < 8; ++i) {
                int r = (idx + i) >> 7, c = (idx + i) & 127;
                float qh = (PR[r][c] - muS[r]) * rsS[r];
                of[(size_t)(rowbase + r) * DD + h * DH + c] = Qc[r][c] + Wln[c] * qh + Bln[c];
            }
        }

        // ---- P6: S[:,slab] -= (Kc*eta)^T[128x16] @ v_new[:,slab]  (K=16 padded)
        //      transposed state: St[cb+n][d] -= D[d][n]
        {
            v16bf bb2;
            #pragma unroll
            for (int e = 0; e < 16; ++e)
                bb2[e] = (koff == 0) ? (bf16_t)VNT[cb + nN][e] : (bf16_t)0.0f;
            #pragma unroll
            for (int dt = 0; dt < 8; ++dt) {
                const int d0 = dt * 16;
                v16bf aa;
                #pragma unroll
                for (int e = 0; e < 16; ++e) {  // lanes<16: KET row (contiguous); pad otherwise
                    int kk = e + koff;
                    aa[e] = (kk < 16) ? (bf16_t)KET[d0 + lM][kk] : (bf16_t)0.0f;
                }
                v8f d = {};
                d = __builtin_amdgcn_wmma_f32_16x16x32_bf16(false, aa, false, bb2, (short)0, d, false, false);
                #pragma unroll
                for (int r = 0; r < 8; ++r) St[cb + nN][d0 + r + mh] -= d[r];  // contiguous
            }
        }
        __syncthreads();
    }
}

// =====================================================================
// gated group-RMSNorm: onb = bf16( (o*sig(gate)) * rsqrt(mean^2) * norm_w )
// =====================================================================
__global__ __launch_bounds__(256) void gate_norm_kernel(
    const float* __restrict__ of, const float* __restrict__ gatef,
    const float* __restrict__ normw, bf16_t* __restrict__ onb)
{
    int gw   = blockIdx.x * 8 + (threadIdx.x >> 5);
    int lane = threadIdx.x & 31;
    int row = gw / HH, h = gw % HH;
    if (row >= MROWS) return;
    size_t base = (size_t)row * DD + h * DH + lane * 4;
    float4 o4 = *(const float4*)(of + base);
    float4 g4 = *(const float4*)(gatef + base);
    float og[4];
    og[0] = o4.x * sigm_(g4.x);
    og[1] = o4.y * sigm_(g4.y);
    og[2] = o4.z * sigm_(g4.z);
    og[3] = o4.w * sigm_(g4.w);
    float ss = og[0]*og[0] + og[1]*og[1] + og[2]*og[2] + og[3]*og[3];
    #pragma unroll
    for (int m = 16; m >= 1; m >>= 1) ss += __shfl_xor(ss, m, 32);
    float r = rsqrtf(ss / DH + EPSF);
    #pragma unroll
    for (int j = 0; j < 4; ++j)
        onb[base + j] = (bf16_t)(og[j] * r * normw[h * DH + lane * 4 + j]);
}

// =====================================================================
// host-side orchestration
// =====================================================================
extern "C" void kernel_launch(void* const* d_in, const int* in_sizes, int n_in,
                              void* d_out, int out_size, void* d_ws, size_t ws_size,
                              hipStream_t stream)
{
    (void)in_sizes; (void)n_in; (void)out_size; (void)ws_size;
    const float* x     = (const float*)d_in[0];
    const float* Wq    = (const float*)d_in[1];
    const float* Wk    = (const float*)d_in[2];
    const float* Wv    = (const float*)d_in[3];
    const float* Wo    = (const float*)d_in[4];
    const float* Wbeta = (const float*)d_in[5];
    const float* ln_w  = (const float*)d_in[6];
    const float* ln_b  = (const float*)d_in[7];
    const float* Wg1   = (const float*)d_in[8];
    const float* Wg2   = (const float*)d_in[9];
    const float* normw = (const float*)d_in[10];
    float* out = (float*)d_out;

    // ---- workspace carve ----
    char* ws = (char*)d_ws;
    size_t off = 0;
    auto carve = [&](size_t bytes) -> char* {
        char* p = ws + off;
        off += (bytes + 255) & ~(size_t)255;
        return p;
    };
    bf16_t* xb    = (bf16_t*)carve((size_t)MROWS * DD * 2);
    bf16_t* wqt   = (bf16_t*)carve((size_t)DD * DD * 2);   // W^T [N][K] bf16
    bf16_t* wkt   = (bf16_t*)carve((size_t)DD * DD * 2);
    bf16_t* wvt   = (bf16_t*)carve((size_t)DD * DD * 2);
    bf16_t* wot   = (bf16_t*)carve((size_t)DD * DD * 2);
    bf16_t* wg1t  = (bf16_t*)carve((size_t)DH * DD * 2);   // [128][2048]
    bf16_t* wg2t  = (bf16_t*)carve((size_t)DD * DH * 2);   // [2048][128]
    float*  qfp   = (float*)carve((size_t)MROWS * DD * 4);
    float*  kfp   = (float*)carve((size_t)MROWS * DD * 4);
    float*  vfp   = (float*)carve((size_t)MROWS * DD * 4);
    float*  etaf  = (float*)carve((size_t)MROWS * HH * 4);
    float*  ofp   = (float*)carve((size_t)MROWS * DD * 4);
    float*  gate1 = (float*)carve((size_t)MROWS * DH * 4);
    float*  gatef = (float*)carve((size_t)MROWS * DD * 4);
    bf16_t* g1b   = (bf16_t*)carve((size_t)MROWS * DH * 2);
    bf16_t* onb   = (bf16_t*)carve((size_t)MROWS * DD * 2);

    auto cvtN = [&](const float* src, bf16_t* dst, int n) {
        cvt_f32_to_bf16<<<(n / 4 + 255) / 256, 256, 0, stream>>>(src, dst, n);
    };
    auto cvtT = [&](const float* src, bf16_t* dstT, int K, int N) {
        dim3 g(N / 32, K / 32), blk(32, 8);
        cvt_transpose_kernel<<<g, blk, 0, stream>>>(src, dstT, K, N);
    };

    // 1) bf16 staging: x row-major, weights transposed
    cvtN(x, xb, MROWS * DD);
    cvtT(Wq, wqt, DD, DD);
    cvtT(Wk, wkt, DD, DD);
    cvtT(Wv, wvt, DD, DD);
    cvtT(Wo, wot, DD, DD);
    cvtT(Wg1, wg1t, DD, DH);
    cvtT(Wg2, wg2t, DH, DD);

    // 2) projections (WMMA GEMMs, async double-buffered staging)
    dim3 gBig(DD / GTN, MROWS / GTM);          // (16, 64)
    gemm_bf16_kernel<<<gBig, 256, 0, stream>>>(xb, wqt, qfp, MROWS, DD, DD, 1);  // silu
    gemm_bf16_kernel<<<gBig, 256, 0, stream>>>(xb, wkt, kfp, MROWS, DD, DD, 0);
    gemm_bf16_kernel<<<gBig, 256, 0, stream>>>(xb, wvt, vfp, MROWS, DD, DD, 0);

    // 3) k l2-normalize per head; eta = 2*sigmoid(x@Wbeta)
    l2norm_heads_kernel<<<(MROWS * HH) / 8, 256, 0, stream>>>(kfp);
    beta_kernel<<<MROWS / 8, 256, 0, stream>>>(x, Wbeta, etaf);

    // 4) TTT chunked dual-form scan (S fp32 transposed in LDS, TDM chunk
    //    staging, WMMA einsums)
    ttt_scan_kernel<<<BB * HH, 256, 0, stream>>>(qfp, kfp, vfp, etaf, ln_w, ln_b, ofp);

    // 5) gate = (x @ Wg1) @ Wg2
    dim3 gG1(DH / GTN, MROWS / GTM);           // (1, 64)
    gemm_bf16_kernel<<<gG1, 256, 0, stream>>>(xb, wg1t, gate1, MROWS, DH, DD, 0);
    cvtN(gate1, g1b, MROWS * DH);
    gemm_bf16_kernel<<<gBig, 256, 0, stream>>>(g1b, wg2t, gatef, MROWS, DD, DH, 0);

    // 6) gated group-RMSNorm -> bf16
    gate_norm_kernel<<<(MROWS * HH) / 8, 256, 0, stream>>>(ofp, gatef, normw, onb);

    // 7) final projection: out = onorm @ Wo (f32 out)
    gemm_bf16_kernel<<<gBig, 256, 0, stream>>>(onb, wot, out, MROWS, DD, DD, 0);
}
